// GraphCast_40114994545133
// MI455X (gfx1250) — matrix operand order repaired and checked by reading, not compiled, not run on hardware
//
#include <hip/hip_runtime.h>

// ---------------------------------------------------------------------------
// GraphCast-style GNN for MI455X (gfx1250, wave32, WMMA + TDM).
// GEMMs: v_wmma_f32_16x16x32_bf16 (f32 accumulate). Activations bf16,
// aggregations f32. Weights are pre-transposed/converted to bf16 once per
// launch; the per-K-step weight slab is then DMA'd into LDS by the Tensor
// Data Mover (tensor_load_to_lds) with hardware padding that reproduces the
// bank-conflict-free stride-40 layout. Fused MLP:
//   out = [res +] LN( W2 @ silu( W1 @ gather_concat(x) + b1 ) + b2 )
// LDS: sA [32][40] bf16 (A tile), sBt [512][40] bf16 (TDM dest),
//      sH [32][520] bf16 (hidden), sY [32][512] f32 (overlay, LayerNorm).
// ---------------------------------------------------------------------------

typedef __attribute__((ext_vector_type(16))) __bf16 v16bf;
typedef __attribute__((ext_vector_type(8)))  __bf16 v8bf;
typedef __attribute__((ext_vector_type(8)))  float  v8f;
typedef __attribute__((ext_vector_type(4)))  unsigned int u32x4;
typedef __attribute__((ext_vector_type(8)))  int          i32x8;
typedef __attribute__((ext_vector_type(4)))  int          i32x4;

#define HID   512
#define MT    32      // rows per block tile
#define BLK   512     // threads per block = 16 wave32s
#define LDA   40      // sA padded stride (bf16 elems)
#define LDB   40      // sBt padded stride (bf16 elems) == TDM 64B row + 16B pad
#define LDH   520     // sH padded stride (bf16 elems)
#define NGRID 100000
#define NMESH 10242
#define EG2M  200000
#define EMESH 81920
#define EM2G  200000

// Input row = concat of up to 3 segments; each segment is either raw f32
// features or a bf16 state, optionally gathered through an index array.
struct Seg { const void* ptr; const int* idx; int width; int is_f32; };

// ---- WMMA fragment loads (layouts per CDNA5 ISA 7.12.2, wave32) -----------
__device__ __forceinline__ v16bf load_a_frag(const __bf16* base, int ld,
                                             int row0, int k0, int lane) {
  const __bf16* p = base + (row0 + (lane & 15)) * ld + k0 + ((lane >> 4) << 3);
  v8bf lo = *(const v8bf*)p;
  v8bf hi = *(const v8bf*)(p + 16);
  return __builtin_shufflevector(lo, hi, 0,1,2,3,4,5,6,7,8,9,10,11,12,13,14,15);
}
__device__ __forceinline__ v16bf load_b_frag(const __bf16* sBt, int ncol, int lane) {
  const __bf16* p = sBt + (ncol + (lane & 15)) * LDB + ((lane >> 4) << 3);
  v8bf lo = *(const v8bf*)p;
  v8bf hi = *(const v8bf*)(p + 16);
  return __builtin_shufflevector(lo, hi, 0,1,2,3,4,5,6,7,8,9,10,11,12,13,14,15);
}

// ---- TDM: DMA a [ncols x 32] bf16 tile of the transposed weight matrix ----
// Wt is [ncols][nk] bf16 row-major. Tile starts at column k0 (32 wide).
// pad_interval=16 DWORDs (64B = one tile row), pad_amount=4 DWORDs (16B)
// -> LDS rows land with stride 80B == LDB bf16 elements.
__device__ __forceinline__ void tdm_load_slab(const __bf16* wt, int k0, int nk,
                                              int ncols, unsigned lds_off) {
  unsigned long long ga = (unsigned long long)wt + (unsigned long long)k0 * 2u;
  u32x4 g0;
  g0[0] = 1u;                                           // count=1, no gather
  g0[1] = lds_off;                                      // LDS byte address
  g0[2] = (unsigned)ga;                                 // global_addr[31:0]
  g0[3] = ((unsigned)(ga >> 32) & 0x01FFFFFFu)          // global_addr[56:32]
          | 0x80000000u;                                // type=2 ("image")
  i32x8 g1;
  g1[0] = (int)((1u << 16)                              // data_size = 2 bytes
              | (1u << 20)                              // pad_enable
              | (3u << 22)                              // pad_interval: 16 DW
              | (3u << 25));                            // pad_amount: 4 DW
  g1[1] = (int)(((unsigned)nk & 0xFFFFu) << 16);        // tensor_dim0[15:0]
  g1[2] = (int)((((unsigned)nk >> 16) & 0xFFFFu)        // tensor_dim0[31:16]
              | (((unsigned)ncols & 0xFFFFu) << 16));   // tensor_dim1[15:0]
  g1[3] = (int)((((unsigned)ncols >> 16) & 0xFFFFu)     // tensor_dim1[31:16]
              | (32u << 16));                           // tile_dim0 = 32
  g1[4] = (int)((unsigned)ncols & 0xFFFFu);             // tile_dim1; tile_dim2=0
  g1[5] = nk;                                           // tensor_dim0_stride lo
  g1[6] = 0;                                            // stride hi / dim1_stride
  g1[7] = 0;
  i32x4 z4 = {0, 0, 0, 0};
  i32x8 z8 = {0, 0, 0, 0, 0, 0, 0, 0};
  __builtin_amdgcn_tensor_load_to_lds(g0, g1, z4, z4, z8, 0);
}

__global__ __launch_bounds__(BLK)
void mlp_fused(Seg s0, Seg s1, Seg s2,
               const __bf16* __restrict__ Wt1, const float* __restrict__ B1, int din,
               const __bf16* __restrict__ Wt2, const float* __restrict__ B2, int dout,
               const float* __restrict__ LNg, const float* __restrict__ LNb, int use_ln,
               const __bf16* __restrict__ res,
               __bf16* __restrict__ out_bf, float* __restrict__ out_f32, int M)
{
  extern __shared__ char smem[];
  __bf16* sBt = (__bf16*)smem;                        // [512][LDB] W slab (TDM dest)
  __bf16* sH  = (__bf16*)(smem + HID * LDB * 2);      // [32][LDH] hidden (bf16)
  float*  sY  = (float*)smem;                         // [32][512] f32, overlays
  __shared__ __bf16 sA[MT * LDA];                     // gathered A tile

  const int tid  = threadIdx.x;
  const int lane = tid & 31;
  const int wave = tid >> 5;
  const int mrow = (wave & 1) << 4;                   // 0 or 16
  const int wN   = (wave >> 1) << 6;                  // 0..448, 64 cols per wave
  const int row0 = blockIdx.x * MT;
  const int w01  = s0.width + s1.width;
  const unsigned sBt_lds = (unsigned)(unsigned long long)sBt;

  // ------------------ phase 1: H = silu(X @ W1 + b1) -----------------------
  v8f acc[4] = {};
  for (int k0 = 0; k0 < din; k0 += 32) {
    __syncthreads();                                  // previous slab reads done
    // stage gathered A tile (MT x 32) in 8-element chunks
    for (int i = tid; i < MT * 4; i += BLK) {
      int r = i >> 2, kc = (i & 3) << 3;
      int gr = row0 + r, kk = k0 + kc;
      __bf16* dstp = &sA[r * LDA + kc];
      bool done = false;
      if (gr < M && kk + 8 <= din) {
        const Seg* s; int c = kk;
        if (c < s0.width)      { s = &s0; }
        else if (c < w01)      { s = &s1; c -= s0.width; }
        else                   { s = &s2; c -= w01; }
        if (c + 8 <= s->width) {
          int sr = s->idx ? s->idx[gr] : gr;
          if (!s->is_f32) {                 // bf16 state: one global_load_b128
            *(v8bf*)dstp = *(const v8bf*)((const __bf16*)s->ptr +
                                          (long)sr * s->width + c);
            done = true;
          } else {                          // f32 (agg): two float4 + convert
            const float* sp = (const float*)s->ptr + (long)sr * s->width + c;
            float4 a0 = *(const float4*)sp;
            float4 a1 = *(const float4*)(sp + 4);
            v8bf t;
            t[0]=(__bf16)a0.x; t[1]=(__bf16)a0.y; t[2]=(__bf16)a0.z; t[3]=(__bf16)a0.w;
            t[4]=(__bf16)a1.x; t[5]=(__bf16)a1.y; t[6]=(__bf16)a1.z; t[7]=(__bf16)a1.w;
            *(v8bf*)dstp = t;
            done = true;
          }
        }
      }
      if (!done) {                          // edges / raw narrow features
        for (int j = 0; j < 8; ++j) {
          int ck = kk + j; float v = 0.f;
          if (gr < M && ck < din) {
            const Seg* s; int c = ck;
            if (c < s0.width)      { s = &s0; }
            else if (c < w01)      { s = &s1; c -= s0.width; }
            else                   { s = &s2; c -= w01; }
            int sr = s->idx ? s->idx[gr] : gr;
            long off = (long)sr * s->width + c;
            v = s->is_f32 ? ((const float*)s->ptr)[off]
                          : (float)((const __bf16*)s->ptr)[off];
          }
          dstp[j] = (__bf16)v;
        }
      }
    }
    // DMA the W1 k-slab into sBt (TDM; OOB columns auto-zero for narrow din)
    if (tid == 0) {
      tdm_load_slab(Wt1, k0, din, HID, sBt_lds);
      __builtin_amdgcn_s_wait_tensorcnt(0);
    }
    __syncthreads();
    v16bf a  = load_a_frag(sA, LDA, mrow, 0, lane);
    v16bf b0 = load_b_frag(sBt, wN +  0, lane);
    v16bf b1 = load_b_frag(sBt, wN + 16, lane);
    v16bf b2 = load_b_frag(sBt, wN + 32, lane);
    v16bf b3 = load_b_frag(sBt, wN + 48, lane);
    acc[0] = __builtin_amdgcn_wmma_f32_16x16x32_bf16(false, a, false, b0, (short)0, acc[0], false, false);
    acc[1] = __builtin_amdgcn_wmma_f32_16x16x32_bf16(false, a, false, b1, (short)0, acc[1], false, false);
    acc[2] = __builtin_amdgcn_wmma_f32_16x16x32_bf16(false, a, false, b2, (short)0, acc[2], false, false);
    acc[3] = __builtin_amdgcn_wmma_f32_16x16x32_bf16(false, a, false, b3, (short)0, acc[3], false, false);
  }
  __syncthreads();
  // bias + SiLU -> sH.  C/D layout: VGPR r holds M = mrow+r+8*(lane>=16), N = lane&15
#pragma unroll
  for (int nt = 0; nt < 4; ++nt) {
    int n = wN + nt * 16 + (lane & 15);
    float bias = B1[n];
#pragma unroll
    for (int r = 0; r < 8; ++r) {
      int mm = mrow + r + ((lane >> 4) << 3);
      float x = acc[nt][r] + bias;
      sH[mm * LDH + n] = (__bf16)(x / (1.f + __expf(-x)));
    }
  }

  // ------------------ phase 2: Y = H @ W2 + b2 -----------------------------
  const int ndp = (dout + 15) & ~15;                  // padded to 16-col tiles
  v8f acc2[4] = {};
  for (int k0 = 0; k0 < HID; k0 += 32) {
    __syncthreads();
    if (tid == 0) {                                   // Wt2 is [dout][512]
      tdm_load_slab(Wt2, k0, HID, dout, sBt_lds);
      __builtin_amdgcn_s_wait_tensorcnt(0);
    }
    __syncthreads();
    if (dout == HID) {                                // common LN case: preload
      v16bf a  = load_a_frag(sH, LDH, mrow, k0, lane);
      v16bf b0 = load_b_frag(sBt, wN +  0, lane);
      v16bf b1 = load_b_frag(sBt, wN + 16, lane);
      v16bf b2 = load_b_frag(sBt, wN + 32, lane);
      v16bf b3 = load_b_frag(sBt, wN + 48, lane);
      acc2[0] = __builtin_amdgcn_wmma_f32_16x16x32_bf16(false, a, false, b0, (short)0, acc2[0], false, false);
      acc2[1] = __builtin_amdgcn_wmma_f32_16x16x32_bf16(false, a, false, b1, (short)0, acc2[1], false, false);
      acc2[2] = __builtin_amdgcn_wmma_f32_16x16x32_bf16(false, a, false, b2, (short)0, acc2[2], false, false);
      acc2[3] = __builtin_amdgcn_wmma_f32_16x16x32_bf16(false, a, false, b3, (short)0, acc2[3], false, false);
    } else if (wN < ndp) {                            // small head (dout=5)
      v16bf a = load_a_frag(sH, LDH, mrow, k0, lane);
#pragma unroll
      for (int nt = 0; nt < 4; ++nt) {
        int nc = wN + nt * 16;
        if (nc < ndp) {
          v16bf b = load_b_frag(sBt, nc, lane);
          acc2[nt] = __builtin_amdgcn_wmma_f32_16x16x32_bf16(
              false, a, false, b, (short)0, acc2[nt], false, false);
        }
      }
    }
  }

  if (!use_ln) {                                      // small head: raw f32 logits
#pragma unroll
    for (int nt = 0; nt < 4; ++nt) {
      int nc = wN + nt * 16;
      if (nc >= ndp) continue;
      int n = nc + (lane & 15);
      float bias = (n < dout) ? B2[n] : 0.f;
#pragma unroll
      for (int r = 0; r < 8; ++r) {
        int mm = mrow + r + ((lane >> 4) << 3);
        int gr = row0 + mm;
        if (gr < M && n < dout)
          out_f32[(long)gr * dout + n] = acc2[nt][r] + bias;
      }
    }
    return;
  }

  // ---- LayerNorm over 512 cols (+ optional residual) ----------------------
  __syncthreads();                                    // sBt/sH dead -> reuse as sY
#pragma unroll
  for (int nt = 0; nt < 4; ++nt) {
    int n = wN + nt * 16 + (lane & 15);
    float bias = B2[n];
#pragma unroll
    for (int r = 0; r < 8; ++r) {
      int mm = mrow + r + ((lane >> 4) << 3);
      sY[mm * HID + n] = acc2[nt][r] + bias;
    }
  }
  __syncthreads();
  {
    int rrow = tid >> 4, l16 = tid & 15;              // 16 threads per row
    float s = 0.f, ss = 0.f;
#pragma unroll
    for (int j = 0; j < HID / 16; ++j) {
      float v = sY[rrow * HID + l16 + j * 16];
      s += v; ss += v * v;
    }
#pragma unroll
    for (int o = 8; o > 0; o >>= 1) {                 // reduce within 16-lane group
      s  += __shfl_xor(s,  o, 16);
      ss += __shfl_xor(ss, o, 16);
    }
    float mu  = s * (1.f / HID);
    float inv = rsqrtf(ss * (1.f / HID) - mu * mu + 1e-5f);
    int gr = row0 + rrow;
    if (gr < M) {
      for (int j = 0; j < HID / 16; ++j) {
        int c = l16 + j * 16;
        float y  = (sY[rrow * HID + c] - mu) * inv * LNg[c] + LNb[c];
        float rv = res ? (float)res[(long)gr * HID + c] : 0.f;
        out_bf[(long)gr * HID + c] = (__bf16)(rv + y);
      }
    }
  }
}

// ---------------------------------------------------------------------------
__global__ void transpose_w_kernel(const float* __restrict__ W,
                                   __bf16* __restrict__ Wt, int K, int N) {
  long i = (long)blockIdx.x * blockDim.x + threadIdx.x;
  if (i >= (long)K * N) return;
  int k = (int)(i / N), n = (int)(i - (long)k * N);
  Wt[(long)n * K + k] = (__bf16)W[i];                 // Wt[N][K]
}

__global__ void zero_kernel(float* p, long n) {
  long i = (long)blockIdx.x * blockDim.x + threadIdx.x;
  if (i < n) p[i] = 0.f;
}

__global__ void scatter_kernel(const __bf16* __restrict__ e,
                               const int* __restrict__ dst,
                               float* __restrict__ agg, int E) {
  long i = (long)blockIdx.x * blockDim.x + threadIdx.x;
  if (i >= (long)E * HID) return;
  int ei = (int)(i >> 9), c = (int)(i & (HID - 1));
  atomicAdd(agg + (long)dst[ei] * HID + c, (float)e[i]);
}

__global__ void softmax_kernel(float* out, int M, int C) {
  int r = blockIdx.x * blockDim.x + threadIdx.x;
  if (r >= M) return;
  float* p = out + (long)r * C;
  float mx = p[0];
  for (int i = 1; i < C; ++i) mx = fmaxf(mx, p[i]);
  float s = 0.f;
  for (int i = 0; i < C; ++i) { float e = __expf(p[i] - mx); p[i] = e; s += e; }
  float inv = 1.f / s;
  for (int i = 0; i < C; ++i) p[i] *= inv;
}

// ---------------------------------------------------------------------------
struct MlpP {
  const float *w1, *b1, *w2, *b2, *ln_g, *ln_b;
  __bf16 *wt1, *wt2;                                  // transposed bf16 weights
};

static inline Seg mkseg(const void* p, const int* idx, int w, int f32) {
  Seg s; s.ptr = p; s.idx = idx; s.width = w; s.is_f32 = f32; return s;
}

static void run_mlp(hipStream_t st, const MlpP& P, Seg s0, Seg s1, Seg s2,
                    int din, int dout, int use_ln, const __bf16* res,
                    __bf16* out_bf, float* out_f, int M) {
  int nb = (M + MT - 1) / MT;
  size_t dyn = (size_t)HID * LDB * 2 + (size_t)MT * LDH * 2;  // sBt + sH (sY overlays)
  mlp_fused<<<nb, BLK, dyn, st>>>(s0, s1, s2, P.wt1, P.b1, din, P.wt2, P.b2, dout,
                                  P.ln_g, P.ln_b, use_ln, res, out_bf, out_f, M);
}

static void run_zero(hipStream_t st, float* p, long n) {
  zero_kernel<<<(int)((n + 255) / 256), 256, 0, st>>>(p, n);
}

static void run_scat(hipStream_t st, const __bf16* e, const int* d, float* a, int E) {
  long n = (long)E * HID;
  scatter_kernel<<<(int)((n + 255) / 256), 256, 0, st>>>(e, d, a, E);
}

extern "C" void kernel_launch(void* const* d_in, const int* in_sizes, int n_in,
                              void* d_out, int out_size, void* d_ws, size_t ws_size,
                              hipStream_t stream) {
  (void)in_sizes; (void)n_in; (void)out_size; (void)ws_size;

  const float* X       = (const float*)d_in[0];   // [NGRID,10,1] == [NGRID,10]
  const float* mesh_x  = (const float*)d_in[1];
  const int*   g2m_src = (const int*)d_in[2];
  const int*   g2m_dst = (const int*)d_in[3];
  const float* g2m_ef  = (const float*)d_in[4];
  const int*   msh_src = (const int*)d_in[5];
  const int*   msh_dst = (const int*)d_in[6];
  const float* msh_ef  = (const float*)d_in[7];
  const int*   m2g_src = (const int*)d_in[8];
  const int*   m2g_dst = (const int*)d_in[9];
  const float* m2g_ef  = (const float*)d_in[10];

  // params flattened in setup_inputs() insertion order
  int pi = 11;
  auto grab = [&](int ln) {
    MlpP p;
    p.w1 = (const float*)d_in[pi++]; p.b1 = (const float*)d_in[pi++];
    p.w2 = (const float*)d_in[pi++]; p.b2 = (const float*)d_in[pi++];
    p.ln_g = ln ? (const float*)d_in[pi++] : nullptr;
    p.ln_b = ln ? (const float*)d_in[pi++] : nullptr;
    p.wt1 = nullptr; p.wt2 = nullptr;
    return p;
  };
  MlpP grid_emb = grab(1), mesh_emb = grab(1), g2m_eemb = grab(1);
  MlpP mesh_eemb = grab(1), m2g_eemb = grab(1);
  MlpP g2m_edge = grab(1), g2m_node = grab(1), g2m_grid = grab(1);
  MlpP m2g_edge = grab(1), m2g_node = grab(1);
  MlpP proc[8];
  for (int i = 0; i < 8; ++i) proc[i] = grab(1);    // 4 x {edge,node}
  MlpP finalp = grab(0);

  // workspace: bf16 states + f32 aggregation buffers + transposed weights
  char* wsp = (char*)d_ws;
  auto take = [&](size_t b) { void* p = (void*)wsp; wsp += (b + 255) & ~(size_t)255; return p; };
  __bf16* g     = (__bf16*)take((size_t)NGRID * HID * 2);
  __bf16* m     = (__bf16*)take((size_t)NMESH * HID * 2);
  __bf16* eg2m  = (__bf16*)take((size_t)EG2M  * HID * 2);
  __bf16* emesh = (__bf16*)take((size_t)EMESH * HID * 2);
  __bf16* em2g  = (__bf16*)take((size_t)EM2G  * HID * 2);
  float*  aggm  = (float*) take((size_t)NMESH * HID * 4);
  float*  aggg  = (float*) take((size_t)NGRID * HID * 4);

  // one-time (per launch) weight transpose + bf16 conversion
  auto prep = [&](MlpP& p, int din, int dout) {
    p.wt1 = (__bf16*)take((size_t)din * HID * 2);     // Wt1 [HID][din]
    p.wt2 = (__bf16*)take((size_t)HID * dout * 2);    // Wt2 [dout][HID]
    long n1 = (long)din * HID;
    transpose_w_kernel<<<(int)((n1 + 255) / 256), 256, 0, stream>>>(p.w1, p.wt1, din, HID);
    long n2 = (long)HID * dout;
    transpose_w_kernel<<<(int)((n2 + 255) / 256), 256, 0, stream>>>(p.w2, p.wt2, HID, dout);
  };
  prep(grid_emb, 10, HID);   prep(mesh_emb, 3, HID);
  prep(g2m_eemb, 4, HID);    prep(mesh_eemb, 4, HID);  prep(m2g_eemb, 4, HID);
  prep(g2m_edge, 3 * HID, HID); prep(g2m_node, 2 * HID, HID); prep(g2m_grid, HID, HID);
  prep(m2g_edge, 3 * HID, HID); prep(m2g_node, 2 * HID, HID);
  for (int i = 0; i < 4; ++i) { prep(proc[2 * i], 3 * HID, HID); prep(proc[2 * i + 1], 2 * HID, HID); }
  prep(finalp, HID, 5);

  Seg z = mkseg(nullptr, nullptr, 0, 0);

  // ---- encoders ----
  run_mlp(stream, grid_emb,  mkseg(X,      nullptr, 10, 1), z, z, 10, HID, 1, nullptr, g,     nullptr, NGRID);
  run_mlp(stream, mesh_emb,  mkseg(mesh_x, nullptr,  3, 1), z, z,  3, HID, 1, nullptr, m,     nullptr, NMESH);
  run_mlp(stream, g2m_eemb,  mkseg(g2m_ef, nullptr,  4, 1), z, z,  4, HID, 1, nullptr, eg2m,  nullptr, EG2M);
  run_mlp(stream, mesh_eemb, mkseg(msh_ef, nullptr,  4, 1), z, z,  4, HID, 1, nullptr, emesh, nullptr, EMESH);
  run_mlp(stream, m2g_eemb,  mkseg(m2g_ef, nullptr,  4, 1), z, z,  4, HID, 1, nullptr, em2g,  nullptr, EM2G);

  // ---- grid2mesh encoder ----
  run_mlp(stream, g2m_edge, mkseg(eg2m, nullptr, HID, 0),
          mkseg(g, g2m_src, HID, 0), mkseg(m, g2m_dst, HID, 0),
          3 * HID, HID, 1, eg2m, eg2m, nullptr, EG2M);
  run_zero(stream, aggm, (long)NMESH * HID);
  run_scat(stream, eg2m, g2m_dst, aggm, EG2M);
  run_mlp(stream, g2m_node, mkseg(m, nullptr, HID, 0), mkseg(aggm, nullptr, HID, 1), z,
          2 * HID, HID, 1, m, m, nullptr, NMESH);
  run_mlp(stream, g2m_grid, mkseg(g, nullptr, HID, 0), z, z,
          HID, HID, 1, g, g, nullptr, NGRID);

  // ---- processor: 4 rounds of mesh message passing ----
  for (int l = 0; l < 4; ++l) {
    run_mlp(stream, proc[2 * l + 0], mkseg(emesh, nullptr, HID, 0),
            mkseg(m, msh_src, HID, 0), mkseg(m, msh_dst, HID, 0),
            3 * HID, HID, 1, emesh, emesh, nullptr, EMESH);
    run_zero(stream, aggm, (long)NMESH * HID);
    run_scat(stream, emesh, msh_dst, aggm, EMESH);
    run_mlp(stream, proc[2 * l + 1], mkseg(m, nullptr, HID, 0),
            mkseg(aggm, nullptr, HID, 1), z, 2 * HID, HID, 1, m, m, nullptr, NMESH);
  }

  // ---- mesh2grid decoder ----
  run_mlp(stream, m2g_edge, mkseg(em2g, nullptr, HID, 0),
          mkseg(m, m2g_src, HID, 0), mkseg(g, m2g_dst, HID, 0),
          3 * HID, HID, 1, em2g, em2g, nullptr, EM2G);
  run_zero(stream, aggg, (long)NGRID * HID);
  run_scat(stream, em2g, m2g_dst, aggg, EM2G);
  run_mlp(stream, m2g_node, mkseg(g, nullptr, HID, 0), mkseg(aggg, nullptr, HID, 1), z,
          2 * HID, HID, 1, g, g, nullptr, NGRID);

  // ---- final head + softmax ----
  run_mlp(stream, finalp, mkseg(g, nullptr, HID, 0), z, z,
          HID, 5, 0, nullptr, nullptr, (float*)d_out, NGRID);
  softmax_kernel<<<(NGRID + 255) / 256, 256, 0, stream>>>((float*)d_out, NGRID, 5);
}
